// LlamaAttention_51539607912
// MI455X (gfx1250) — matrix operand order repaired and checked by reading, not compiled, not run on hardware
//
#include <hip/hip_runtime.h>
#include <hip/hip_bf16.h>

// ---------------------------------------------------------------------------
// Llama attention block for gfx1250 (MI455X), bf16 WMMA everywhere.
//   B=4, S=2048, HID=1024, H=16, d=64
// Pipeline: convert/transposes -> 3 projection GEMMs -> RoPE -> flash attn
//           -> output GEMM (+ seq_len masking).
// ---------------------------------------------------------------------------

#define B_  4
#define S_  2048
#define HID_ 1024
#define H_  16
#define D_  64

typedef __attribute__((ext_vector_type(16))) __bf16     v16bf;
typedef __attribute__((ext_vector_type(8)))  float      v8f;
typedef __attribute__((ext_vector_type(4)))  unsigned   u32x4;

union Frag {
    v16bf v;
    u32x4 u[2];
    unsigned short s[16];
};

__device__ __forceinline__ unsigned short f2bf(float f) {
    union { float f; unsigned u; } x; x.f = f;
    unsigned r = x.u + 0x7FFFu + ((x.u >> 16) & 1u);   // round-to-nearest-even
    return (unsigned short)(r >> 16);
}
__device__ __forceinline__ float bf2f(unsigned short h) {
    union { unsigned u; float f; } x; x.u = ((unsigned)h) << 16; return x.f;
}

__device__ __forceinline__ v8f wmma_bf16(const Frag& a, const Frag& b, v8f c) {
    // D = A(16x32 bf16) * B(32x16 bf16) + C(16x16 f32)
    return __builtin_amdgcn_wmma_f32_16x16x32_bf16(
        false, a.v, false, b.v, (short)0, c, false, false);
}

// ---------------------------------------------------------------------------
// Conversion kernels
// ---------------------------------------------------------------------------
__global__ void conv_f32_to_bf16(const float* __restrict__ in,
                                 unsigned short* __restrict__ out, int n) {
    int i = blockIdx.x * blockDim.x + threadIdx.x;
    if (i < n) out[i] = f2bf(in[i]);
}

// W is [K][N] row-major f32; produce Wt[N][K] bf16 so B-fragments load contiguously.
__global__ void conv_w_transpose(const float* __restrict__ W,
                                 unsigned short* __restrict__ Wt) {
    int i = blockIdx.x * blockDim.x + threadIdx.x;   // i = n*1024 + k
    int nn = i >> 10, k = i & 1023;
    Wt[i] = f2bf(W[k * HID_ + nn]);
}

// ---------------------------------------------------------------------------
// bf16 GEMM: C[M=8192][N=1024] = A[M][K=1024] * Bt[N][K]^T
// One wave -> 32x64 C tile (2x4 grid of 16x16 WMMA accumulators).
// mode 0: store bf16 row-major          (Q, K projections)
// mode 1: store bf16 as Vt[B][H][d][S]  (V projection, pre-transposed for PV)
// mode 2: store f32 row-major, zeroing rows with s >= seq_lens[b]  (output)
// ---------------------------------------------------------------------------
__global__ void gemm_bf16_kernel(const unsigned short* __restrict__ A,
                                 const unsigned short* __restrict__ Bt,
                                 void* __restrict__ Out,
                                 const int* __restrict__ seq_lens,
                                 int mode) {
    const int wave = threadIdx.x >> 5;
    const int lane = threadIdx.x & 31;
    const int g = lane >> 4;        // lane group (0/1)
    const int n = lane & 15;

    const int m0 = blockIdx.x * 32;
    const int n0 = blockIdx.y * 256 + wave * 64;

    v8f acc[2][4];
    const v8f vzero = {};
    for (int r = 0; r < 2; ++r)
        for (int c = 0; c < 4; ++c) acc[r][c] = vzero;

    for (int k0 = 0; k0 < HID_; k0 += 32) {
        Frag fa[2];
#pragma unroll
        for (int r = 0; r < 2; ++r) {
            int row = m0 + r * 16 + n;
            const u32x4* ap = (const u32x4*)(A + (size_t)row * HID_ + k0);
            fa[r].u[0] = ap[g];          // K = g*8 .. g*8+7
            fa[r].u[1] = ap[2 + g];      // K = 16+g*8 .. 16+g*8+7
        }
        Frag fb[4];
#pragma unroll
        for (int c = 0; c < 4; ++c) {
            int col = n0 + c * 16 + n;
            const u32x4* bp = (const u32x4*)(Bt + (size_t)col * HID_ + k0);
            fb[c].u[0] = bp[g];          // K = g*8 .. g*8+7
            fb[c].u[1] = bp[2 + g];      // K = 16+g*8 .. 16+g*8+7
        }
#pragma unroll
        for (int r = 0; r < 2; ++r)
#pragma unroll
            for (int c = 0; c < 4; ++c)
                acc[r][c] = wmma_bf16(fa[r], fb[c], acc[r][c]);
    }

#pragma unroll
    for (int r = 0; r < 2; ++r)
#pragma unroll
        for (int c = 0; c < 4; ++c)
#pragma unroll
            for (int v = 0; v < 8; ++v) {
                int row = m0 + r * 16 + v + 8 * g;   // C layout: M = v + 8*g
                int col = n0 + c * 16 + n;           //           N = lane&15
                float x = acc[r][c][v];
                if (mode == 0) {
                    ((unsigned short*)Out)[(size_t)row * HID_ + col] = f2bf(x);
                } else if (mode == 1) {
                    int bb = row >> 11, s = row & (S_ - 1);
                    int hh = col >> 6,  dd = col & (D_ - 1);
                    ((unsigned short*)Out)[(((size_t)(bb * H_ + hh) * D_ + dd) << 11) + s] = f2bf(x);
                } else {
                    int bb = row >> 11, s = row & (S_ - 1);
                    ((float*)Out)[(size_t)row * HID_ + col] =
                        (s < seq_lens[bb]) ? x : 0.0f;
                }
            }
}

// ---------------------------------------------------------------------------
// RoPE (in-place on bf16 [B,S,HID]); one thread handles a (j, j+32) pair.
// ---------------------------------------------------------------------------
__global__ void rope_kernel(unsigned short* __restrict__ buf) {
    int i = blockIdx.x * blockDim.x + threadIdx.x;   // B*S*H*32 threads
    int j = i & 31;
    int h = (i >> 5) & 15;
    int s = (i >> 9) & (S_ - 1);
    int b = i >> 20;
    size_t base = (((size_t)(b * S_ + s)) << 10) + h * D_ + j;
    // inv_freq = 10000^(-j/32) = exp(-j * ln(10000)/32)
    float inv = __expf(-(float)j * 0.28782313662425574f);
    float ang = (float)s * inv;
    float cs = cosf(ang), sn = sinf(ang);
    float x1 = bf2f(buf[base]);
    float x2 = bf2f(buf[base + 32]);
    buf[base]      = f2bf(x1 * cs - x2 * sn);
    buf[base + 32] = f2bf(x2 * cs + x1 * sn);
}

// ---------------------------------------------------------------------------
// Flash attention: one wave per (b, h, 16-query tile); 32-key chunks.
//   scores: 2 WMMAs per 16-key tile (d=64 -> two K=32 slices)
//   P re-layout C->A via per-wave LDS (s_wait_dscnt for wave-local ordering)
//   PV: 4 WMMAs per chunk into fp32 O accumulators
// ---------------------------------------------------------------------------
__global__ void attn_kernel(const unsigned short* __restrict__ Qb,
                            const unsigned short* __restrict__ Kb,
                            const unsigned short* __restrict__ Vt,
                            unsigned short* __restrict__ attnb,
                            const int* __restrict__ seq_lens) {
    __shared__ __align__(16) unsigned short plds[4][16 * 32];

    const int bh = blockIdx.x;
    const int b  = bh / H_;
    const int h  = bh % H_;
    const int wave = threadIdx.x >> 5;
    const int lane = threadIdx.x & 31;
    const int g = lane >> 4;
    const int n = lane & 15;

    const int q0 = (blockIdx.y * 4 + wave) * 16;
    const int slen = seq_lens[b];
    const float scale = 0.125f;   // 1/sqrt(64)

    // Q A-fragments (reused for every key chunk): two K=32 slices of d=64.
    Frag qf[2];
    {
        const int qrow = q0 + n;
        const u32x4* qp = (const u32x4*)(Qb + (size_t)(b * S_ + qrow) * HID_ + h * D_);
        qf[0].u[0] = qp[g];     qf[0].u[1] = qp[2 + g];
        qf[1].u[0] = qp[4 + g]; qf[1].u[1] = qp[6 + g];
    }

    v8f O[4];
    const v8f vzero = {};
    for (int t = 0; t < 4; ++t) O[t] = vzero;
    float Mx[8], Ls[8];
#pragma unroll
    for (int v = 0; v < 8; ++v) { Mx[v] = -1e30f; Ls[v] = 0.0f; }

    unsigned short* pl = &plds[wave][0];
    const int kb_end = q0 + 15;          // causal: keys beyond last query useless

    for (int kb = 0; kb <= kb_end; kb += 32) {
        // ---- scores for two 16-key tiles -------------------------------
        v8f sacc[2];
#pragma unroll
        for (int t = 0; t < 2; ++t) {
            int key = kb + t * 16 + n;
            int keyc = key < (S_ - 1) ? key : (S_ - 1);  // clamp (masked later)
            const u32x4* kp = (const u32x4*)(Kb + (size_t)(b * S_ + keyc) * HID_ + h * D_);
            Frag kf0, kf1;
            kf0.u[0] = kp[g]; kf0.u[1] = kp[2 + g];          // d-slice 0..31
            kf1.u[0] = kp[4 + g]; kf1.u[1] = kp[6 + g];          // d-slice 32..63
            v8f z = vzero;
            z = wmma_bf16(qf[0], kf0, z);
            z = wmma_bf16(qf[1], kf1, z);
            sacc[t] = z;
        }

        // ---- online softmax update -------------------------------------
        float p0v[8], p1v[8], alpha[8];
#pragma unroll
        for (int v = 0; v < 8; ++v) {
            int m = v + 8 * g;
            int q = q0 + m;
            int key0 = kb + n, key1 = kb + 16 + n;
            bool ok0 = (key0 <= q) && (key0 < slen);
            bool ok1 = (key1 <= q) && (key1 < slen);
            float s0 = ok0 ? sacc[0][v] * scale : -1e30f;
            float s1 = ok1 ? sacc[1][v] * scale : -1e30f;
            float rm = fmaxf(s0, s1);
            rm = fmaxf(rm, __shfl_xor(rm, 1));
            rm = fmaxf(rm, __shfl_xor(rm, 2));
            rm = fmaxf(rm, __shfl_xor(rm, 4));
            rm = fmaxf(rm, __shfl_xor(rm, 8));
            float Mn = fmaxf(Mx[v], rm);
            float a  = __expf(Mx[v] - Mn);
            Mx[v] = Mn;
            float e0 = ok0 ? __expf(s0 - Mn) : 0.0f;     // exact zeros for masked
            float e1 = ok1 ? __expf(s1 - Mn) : 0.0f;
            float rs = e0 + e1;
            rs += __shfl_xor(rs, 1);
            rs += __shfl_xor(rs, 2);
            rs += __shfl_xor(rs, 4);
            rs += __shfl_xor(rs, 8);
            Ls[v] = Ls[v] * a + rs;
            alpha[v] = a;
            p0v[v] = e0; p1v[v] = e1;
        }

        // rescale O accumulators (row mapping identical to score C layout)
#pragma unroll
        for (int t = 0; t < 4; ++t)
#pragma unroll
            for (int v = 0; v < 8; ++v) O[t][v] *= alpha[v];

        // ---- P: C layout -> A layout via per-wave LDS ------------------
#pragma unroll
        for (int v = 0; v < 8; ++v) {
            int m = v + 8 * g;
            pl[m * 32 + n]      = f2bf(p0v[v]);
            pl[m * 32 + 16 + n] = f2bf(p1v[v]);
        }
        __builtin_amdgcn_wave_barrier();
        asm volatile("s_wait_dscnt 0" ::: "memory");     // wave-local LDS RAW

        Frag pf;
        {
            const u32x4* pp = (const u32x4*)pl + n * 4;  // row m = lane&15
            pf.u[0] = pp[2 * g];
            pf.u[1] = pp[2 * g + 1];
        }
        __builtin_amdgcn_wave_barrier();

        // ---- PV: accumulate into O (Vt is [B,H,d,S] -> contiguous keys) -
#pragma unroll
        for (int t = 0; t < 4; ++t) {
            int vrow = (b * H_ + h) * D_ + t * 16 + n;
            int kstart = kb + g * 16;
            if (kstart > S_ - 16) kstart = S_ - 16;      // clamp, p==0 there
            const u32x4* vp = (const u32x4*)(Vt + (size_t)vrow * S_ + kstart);
            Frag vf;
            vf.u[0] = vp[0]; vf.u[1] = vp[1];
            O[t] = wmma_bf16(pf, vf, O[t]);
        }
    }

    // ---- normalize and store bf16 [B,S,HID] ----------------------------
#pragma unroll
    for (int t = 0; t < 4; ++t)
#pragma unroll
        for (int v = 0; v < 8; ++v) {
            int m = v + 8 * g;
            float l = Ls[v] > 1e-30f ? Ls[v] : 1e-30f;
            float val = O[t][v] / l;
            attnb[(size_t)(b * S_ + q0 + m) * HID_ + h * D_ + t * 16 + n] = f2bf(val);
        }
}

// ---------------------------------------------------------------------------
// Host-side launch
// ---------------------------------------------------------------------------
extern "C" void kernel_launch(void* const* d_in, const int* in_sizes, int n_in,
                              void* d_out, int out_size, void* d_ws, size_t ws_size,
                              hipStream_t stream) {
    (void)in_sizes; (void)n_in; (void)out_size; (void)ws_size;

    const float* x  = (const float*)d_in[0];
    const float* Wq = (const float*)d_in[1];
    const float* Wk = (const float*)d_in[2];
    const float* Wv = (const float*)d_in[3];
    const float* Wo = (const float*)d_in[4];
    const int* seq  = (const int*)d_in[5];

    char* ws = (char*)d_ws;
    const size_t XN  = (size_t)B_ * S_ * HID_;          // 8,388,608 elements
    unsigned short* xb    = (unsigned short*)(ws);                        // 16 MB
    unsigned short* Wqt   = (unsigned short*)(ws + 16777216);             //  2 MB
    unsigned short* Wkt   = (unsigned short*)(ws + 18874368);
    unsigned short* Wvt   = (unsigned short*)(ws + 20971520);
    unsigned short* Wot   = (unsigned short*)(ws + 23068672);
    unsigned short* Qb    = (unsigned short*)(ws + 25165824);             // 16 MB
    unsigned short* Kb    = (unsigned short*)(ws + 41943040);
    unsigned short* Vt    = (unsigned short*)(ws + 58720256);
    unsigned short* attnb = (unsigned short*)(ws + 75497472);
    float* out = (float*)d_out;

    // 1. conversions
    conv_f32_to_bf16<<<(int)(XN / 256), 256, 0, stream>>>(x, xb, (int)XN);
    conv_w_transpose<<<(HID_ * HID_) / 256, 256, 0, stream>>>(Wq, Wqt);
    conv_w_transpose<<<(HID_ * HID_) / 256, 256, 0, stream>>>(Wk, Wkt);
    conv_w_transpose<<<(HID_ * HID_) / 256, 256, 0, stream>>>(Wv, Wvt);
    conv_w_transpose<<<(HID_ * HID_) / 256, 256, 0, stream>>>(Wo, Wot);

    // 2. projections (M=8192 rows, N=1024 cols; wave tile 32x64, block 4 waves)
    dim3 ggrid(B_ * S_ / 32, HID_ / 256);
    gemm_bf16_kernel<<<ggrid, 128, 0, stream>>>(xb, Wqt, Qb, seq, 0);
    gemm_bf16_kernel<<<ggrid, 128, 0, stream>>>(xb, Wkt, Kb, seq, 0);
    gemm_bf16_kernel<<<ggrid, 128, 0, stream>>>(xb, Wvt, Vt, seq, 1);

    // 3. RoPE on Q and K
    const int RN = B_ * S_ * H_ * 32;                   // 4,194,304 pairs
    rope_kernel<<<RN / 256, 256, 0, stream>>>(Qb);
    rope_kernel<<<RN / 256, 256, 0, stream>>>(Kb);

    // 4. flash attention: grid (B*H, S/64 q-tile groups), 4 waves/block
    attn_kernel<<<dim3(B_ * H_, S_ / 64), 128, 0, stream>>>(Qb, Kb, Vt, attnb, seq);

    // 5. output projection + seq_len masking, f32 out
    gemm_bf16_kernel<<<ggrid, 128, 0, stream>>>(attnb, Wot, out, seq, 2);
}
